// MPNNLayer_34505767256484
// MI455X (gfx1250) — matrix operand order repaired and checked by reading, not compiled, not run on hardware
//
#include <hip/hip_runtime.h>

typedef _Float16 half_t;
typedef _Float16 v8h  __attribute__((ext_vector_type(8)));
typedef _Float16 v16h __attribute__((ext_vector_type(16)));
typedef float    v8f  __attribute__((ext_vector_type(8)));

#define DEVINL __device__ __forceinline__

// ---------------- problem dims ----------------
constexpr int KNBR   = 30;
constexpr int NODES  = 4 * 2048;     // B*N = 8192
constexpr int HV_DIM = 148;          // 3*16 + 100
constexpr int HM_DIM = 183;          // 3*17 + 132

// ---------------- weight fragment table ----------------
// Each WMMA B-fragment: 32(K) x 16(N) f16 = 512 elements = 1KB.
// Storage: frag[lane][i] contiguous 16 f16 per lane; element (k,n):
//   n = nt*16 + (lane&15),  k = kt*32 + (lane>>4)*16 + i
// Fragment index within a matrix: nt*Ktiles + kt.
constexpr int FRAG = 512;
constexpr int OFF_WH1   = 0;                 // 33x33  -> 2x3
constexpr int OFF_WS1   = OFF_WH1 + 6;       // 265x100-> 9x7
constexpr int OFF_WV1   = OFF_WS1 + 63;      // 33x16  -> 2x1
constexpr int OFF_WH2   = OFF_WV1 + 2;       // 16x16  -> 1x1
constexpr int OFF_WS2   = OFF_WH2 + 1;       // 116x100-> 4x7
constexpr int OFF_WV2   = OFF_WS2 + 28;      // 16x16  -> 1x1
constexpr int OFF_WH3   = OFF_WV2 + 1;
constexpr int OFF_WS3   = OFF_WH3 + 1;
constexpr int OFF_WV3   = OFF_WS3 + 28;
constexpr int OFF_DH1WS = OFF_WV3 + 1;       // 132x400-> 5x25
constexpr int OFF_DH2WS = OFF_DH1WS + 125;   // 432x100-> 14x7
constexpr int TOTAL_FRAGS = OFF_DH2WS + 98;  // 354
constexpr size_t FRAG_BYTES = (size_t)TOTAL_FRAGS * FRAG * sizeof(half_t);

// ---------------- weight prep: f32 -> f16 WMMA B-fragment layout ----------------
__global__ void prep_w(const float* __restrict__ w, half_t* __restrict__ dst,
                       int K, int N, int Ktiles, int total) {
  int t = blockIdx.x * blockDim.x + threadIdx.x;
  if (t >= total) return;
  int f = t >> 9, j = t & 511;
  int lane = j >> 4, i = j & 15;
  int nt = f / Ktiles, kt = f % Ktiles;
  int n = nt * 16 + (lane & 15);
  int k = kt * 32 + (lane >> 4) * 16 + i;
  float v = (k < K && n < N) ? w[(size_t)k * N + n] : 0.f;
  dst[t] = (half_t)v;
}

// ---------------- WMMA helpers ----------------
DEVINL v16h load_b(const half_t* p) {           // global, pre-swizzled
  return *(const v16h*)(p + (threadIdx.x & 31) * 16);
}
// A fragment from LDS row-major staging: row m = lane&15, K-halves per (lane>>4)
DEVINL v16h load_a(const half_t* buf, int stride, int kcol) {
  int l = threadIdx.x & 31;
  int base = (l & 15) * stride + kcol + 16 * (l >> 4);
  union { v16h v; v8h p[2]; } u;
  u.p[0] = *(const v8h*)(buf + base);
  u.p[1] = *(const v8h*)(buf + base + 8);
  return u.v;
}
DEVINL v8f wmma16(v16h a, v16h b, v8f c) {
  return __builtin_amdgcn_wmma_f32_16x16x32_f16(false, a, false, b, (short)0, c, false, false);
}

// ---------------- fused GVP on a 16-edge tile (wave-private LDS) ----------------
// SB: [16][288] f16 scalar staging (s_in at cols [0,NS_IN), vn written at [NS_IN,NS_IN+H))
// VB: [3][16][64] f16 vector staging (input v; overwritten with vh for the v_out GEMM)
template<int NS_IN, int H, int KV_T, int KS_T, int NH_T, int KVO_T, bool ACT>
DEVINL void gvp_edge(half_t (&SB)[16][288], half_t (&VB)[3][16][64],
                     const half_t* whF, const half_t* wsF, const half_t* wvF,
                     const float* __restrict__ bs,
                     v8f (&vout)[3], v8f (&sout)[7]) {
  const int l = threadIdx.x & 31, lm = l & 15, hh = l >> 4;
  // 1) vh = v @ wh   (read all of VB before overwriting it)
  v8f vh[3][NH_T];
#pragma unroll
  for (int j = 0; j < NH_T; ++j)
#pragma unroll
    for (int c = 0; c < 3; ++c) {
      v8f acc = {};
#pragma unroll
      for (int kt = 0; kt < KV_T; ++kt)
        acc = wmma16(load_a(&VB[c][0][0], 64, kt * 32),
                     load_b(whF + (j * KV_T + kt) * FRAG), acc);
      vh[c][j] = acc;
    }
  // 2) vn = ||vh||_c  -> SB;  vh -> VB (f16) for the v_out GEMM
#pragma unroll
  for (int j = 0; j < NH_T; ++j) {
    int nch = j * 16 + lm;
#pragma unroll
    for (int r = 0; r < 8; ++r) {
      float x = vh[0][j][r], y = vh[1][j][r], z = vh[2][j][r];
      float vn = sqrtf(fmaxf(x * x + y * y + z * z, 1e-8f));
      int mr = r + 8 * hh;
      if (nch < H) {
        SB[mr][NS_IN + nch] = (half_t)vn;
        VB[0][mr][nch] = (half_t)x;
        VB[1][mr][nch] = (half_t)y;
        VB[2][mr][nch] = (half_t)z;
      }
    }
  }
  // 3) s_out = [s|vn] @ ws + bs  (+relu)
#pragma unroll
  for (int nt = 0; nt < 7; ++nt) {
    v8f acc = {};
#pragma unroll
    for (int kt = 0; kt < KS_T; ++kt)
      acc = wmma16(load_a(&SB[0][0], 288, kt * 32),
                   load_b(wsF + (nt * KS_T + kt) * FRAG), acc);
    int n = nt * 16 + lm;
    float bv = (n < 100) ? bs[n] : 0.f;
#pragma unroll
    for (int r = 0; r < 8; ++r) {
      float v = acc[r] + bv;
      if (ACT) v = fmaxf(v, 0.f);
      acc[r] = v;
    }
    sout[nt] = acc;
  }
  // 4) v_out = vh @ wv  (+sigmoid(||.||) gate) -- norms align lane-for-lane in D regs
#pragma unroll
  for (int c = 0; c < 3; ++c) {
    v8f acc = {};
#pragma unroll
    for (int kt = 0; kt < KVO_T; ++kt)
      acc = wmma16(load_a(&VB[c][0][0], 64, kt * 32), load_b(wvF + kt * FRAG), acc);
    vout[c] = acc;
  }
  if (ACT) {
#pragma unroll
    for (int r = 0; r < 8; ++r) {
      float x = vout[0][r], y = vout[1][r], z = vout[2][r];
      float nrm = sqrtf(fmaxf(x * x + y * y + z * z, 1e-8f));
      float g = 1.f / (1.f + __expf(-nrm));
      vout[0][r] = x * g; vout[1][r] = y * g; vout[2][r] = z * g;
    }
  }
}

// stage this GVP's outputs as the next GVP's inputs
DEVINL void stage_next(half_t (&SB)[16][288], half_t (&VB)[3][16][64],
                       const v8f (&vo)[3], const v8f (&so)[7]) {
  const int l = threadIdx.x & 31, lm = l & 15, hh = l >> 4;
#pragma unroll
  for (int r = 0; r < 8; ++r) {
    int mr = r + 8 * hh;
#pragma unroll
    for (int nt = 0; nt < 7; ++nt) {
      int n = nt * 16 + lm;
      if (n < 100) SB[mr][n] = (half_t)so[nt][r];
    }
    if (lm < 12) SB[mr][116 + lm] = (half_t)0.f;   // zero K-pad 116..127
    VB[0][mr][lm] = (half_t)vo[0][r];
    VB[1][mr][lm] = (half_t)vo[1][r];
    VB[2][mr][lm] = (half_t)vo[2][r];
    VB[0][mr][16 + lm] = (half_t)0.f;              // zero K-pad 16..31
    VB[1][mr][16 + lm] = (half_t)0.f;
    VB[2][mr][16 + lm] = (half_t)0.f;
  }
}

// ---------------- edge kernel: fused GVP1-3 + masked neighbor-mean ----------------
// block = 4 waves = 64 edge slots; handles 2 nodes x 30 neighbors (60 real slots)
__global__ __launch_bounds__(128)
void edge_kernel(const float* __restrict__ hV, const float* __restrict__ hM,
                 const int* __restrict__ maskA, const half_t* __restrict__ wf,
                 const float* __restrict__ bs1, const float* __restrict__ bs2,
                 const float* __restrict__ bs3, float* __restrict__ dhOut) {
  __shared__ __align__(16) half_t SB[4][16][288];
  __shared__ __align__(16) half_t VB[4][3][16][64];
  __shared__ __align__(16) float HVB[2][HV_DIM];
  __shared__ __align__(16) float RB[2][HV_DIM];
  const int tid = threadIdx.x;
  const int w = tid >> 5, l = tid & 31, lm = l & 15, hh = l >> 4;
  const int blk = blockIdx.x;

  {  // zero this wave's staging buffers
    uint4 z = make_uint4(0, 0, 0, 0);
    uint4* ps = (uint4*)&SB[w][0][0];
    for (int i = l; i < 16 * 288 * 2 / 16; i += 32) ps[i] = z;
    uint4* pv = (uint4*)&VB[w][0][0][0];
    for (int i = l; i < 3 * 16 * 64 * 2 / 16; i += 32) pv[i] = z;
  }
  for (int i = tid; i < 2 * HV_DIM; i += 128) {  // stage the block's 2 node rows
    HVB[i / HV_DIM][i % HV_DIM] = hV[(size_t)(blk * 2 + i / HV_DIM) * HV_DIM + i % HV_DIM];
    RB[i / HV_DIM][i % HV_DIM] = 0.f;
  }
  __syncthreads();

  // stage h_EV for this wave's 16 edges (f32 -> f16, concat layout)
#pragma unroll 1
  for (int me = 0; me < 16; ++me) {
    int slot = w * 16 + me;
    if (slot >= 60) break;                       // uniform per wave
    int nl = slot / 30;
    const float* hm = hM + (size_t)(blk * 60 + slot) * HM_DIM;
    for (int i = l; i < HM_DIM; i += 32) {
      float v = hm[i];
      if (i < 51) VB[w][i / 17][me][16 + i % 17] = (half_t)v;   // v ch 16..32
      else        SB[w][me][100 + (i - 51)]      = (half_t)v;   // s 100..231
    }
    for (int i = l; i < HV_DIM; i += 32) {
      float v = HVB[nl][i];
      if (i < 48) VB[w][i / 16][me][i % 16] = (half_t)v;        // v ch 0..15
      else        SB[w][me][i - 48]         = (half_t)v;        // s 0..99
    }
  }

  float scale[8];
#pragma unroll
  for (int r = 0; r < 8; ++r) {
    int slot = w * 16 + r + 8 * hh;
    scale[r] = (slot < 60) ? (float)maskA[blk * 60 + slot] * (1.f / 30.f) : 0.f;
  }

  v8f vo[3], so[7];
  gvp_edge<232, 33, 2, 9, 3, 2, true >(SB[w], VB[w], wf + OFF_WH1 * FRAG, wf + OFF_WS1 * FRAG, wf + OFF_WV1 * FRAG, bs1, vo, so);
  stage_next(SB[w], VB[w], vo, so);
  gvp_edge<100, 16, 1, 4, 1, 1, true >(SB[w], VB[w], wf + OFF_WH2 * FRAG, wf + OFF_WS2 * FRAG, wf + OFF_WV2 * FRAG, bs2, vo, so);
  stage_next(SB[w], VB[w], vo, so);
  gvp_edge<100, 16, 1, 4, 1, 1, false>(SB[w], VB[w], wf + OFF_WH3 * FRAG, wf + OFF_WS3 * FRAG, wf + OFF_WV3 * FRAG, bs3, vo, so);

  // masked mean over neighbors, reduced in LDS
#pragma unroll
  for (int r = 0; r < 8; ++r) {
    int mr = r + 8 * hh, slot = w * 16 + mr;
    if (slot < 60) {
      int nl = slot / 30; float s = scale[r];
      atomicAdd(&RB[nl][ 0 + lm], vo[0][r] * s);
      atomicAdd(&RB[nl][16 + lm], vo[1][r] * s);
      atomicAdd(&RB[nl][32 + lm], vo[2][r] * s);
    }
  }
#pragma unroll
  for (int nt = 0; nt < 7; ++nt) {
    int n = nt * 16 + lm;
    if (n < 100) {
#pragma unroll
      for (int r = 0; r < 8; ++r) {
        int slot = w * 16 + r + 8 * hh;
        if (slot < 60) atomicAdd(&RB[slot / 30][48 + n], so[nt][r] * scale[r]);
      }
    }
  }
  __syncthreads();
  for (int i = tid; i < 2 * HV_DIM; i += 128)
    dhOut[(size_t)(blk * 2 + i / HV_DIM) * HV_DIM + i % HV_DIM] = RB[i / HV_DIM][i % HV_DIM];
}

// ---------------- node kernel: LN1 -> W_dh (WMMA for s-GEMMs) -> LN2 -> mask ----------------
__global__ __launch_bounds__(32)
void node_kernel(const float* __restrict__ hV, const float* __restrict__ dhBuf,
                 const int* __restrict__ maskV, const half_t* __restrict__ wf,
                 const float* __restrict__ w1h, const float* __restrict__ b1dh,
                 const float* __restrict__ w1v, const float* __restrict__ w2h,
                 const float* __restrict__ b2dh, const float* __restrict__ w2v,
                 const float* __restrict__ g1, const float* __restrict__ b1,
                 const float* __restrict__ g2, const float* __restrict__ b2,
                 float* __restrict__ out) {
  __shared__ __align__(16) half_t SB2[16][608];   // [0,160): dh1 s_cat; [160,608): dh2 s_cat
  __shared__ __align__(16) float HVB2[16][HV_DIM];
  __shared__ __align__(16) float VV[3][16][32];
  __shared__ __align__(16) float VH[3][16][32];
  __shared__ __align__(16) float DB[16][HV_DIM];
  const int l = threadIdx.x, lm = l & 15, hh = l >> 4;
  const int blk = blockIdx.x;

  { uint4 z = make_uint4(0, 0, 0, 0);
    uint4* p = (uint4*)&SB2[0][0];
    for (int i = l; i < 16 * 608 * 2 / 16; i += 32) p[i] = z; }

  for (int i = l; i < 16 * HV_DIM; i += 32) {     // x = h_V + dh
    int row = i / HV_DIM, col = i % HV_DIM;
    size_t node = (size_t)blk * 16 + row;
    HVB2[row][col] = hV[node * HV_DIM + col] + dhBuf[node * HV_DIM + col];
  }
  {  // LN1: 2 lanes per node
    int row = l >> 1, part = l & 1;
    float sum = 0.f, sq = 0.f;
    for (int j = part * 50; j < part * 50 + 50; ++j) { float v = HVB2[row][48 + j]; sum += v; sq += v * v; }
    sum += __shfl_xor(sum, 1, 32); sq += __shfl_xor(sq, 1, 32);
    float mu = sum * 0.01f, var = sq * 0.01f - mu * mu, rstd = rsqrtf(var + 1e-3f);
    float vs = 0.f;
    for (int ch = part * 8; ch < part * 8 + 8; ++ch) {
      float q = 0.f;
      for (int c = 0; c < 3; ++c) { float v = HVB2[row][c * 16 + ch]; q += v * v; }
      vs += fmaxf(q, 1e-8f);
    }
    vs += __shfl_xor(vs, 1, 32);
    float vsc = rsqrtf(vs * (1.f / 16.f));
    for (int j = part * 50; j < part * 50 + 50; ++j) {
      float y = (HVB2[row][48 + j] - mu) * rstd * g1[j] + b1[j];
      HVB2[row][48 + j] = y; SB2[row][j] = (half_t)y;
    }
    for (int col = part * 24; col < part * 24 + 24; ++col) {
      float y = HVB2[row][col] * vsc;
      HVB2[row][col] = y; VV[col / 16][row][col & 15] = y;
    }
  }
  {  // dh1 vh: 16->32 per dim (VALU), vn -> SB2[100..131]
    float acc[48];
    for (int t = 0; t < 48; ++t) {
      int o = t * 32 + l, c = o / 512, row = (o >> 5) & 15, hch = o & 31;
      float a = 0.f;
      for (int k2 = 0; k2 < 16; ++k2) a += VV[c][row][k2] * w1h[k2 * 32 + hch];
      acc[t] = a;
    }
    for (int t = 0; t < 48; ++t) {
      int o = t * 32 + l; VH[o / 512][(o >> 5) & 15][o & 31] = acc[t];
    }
    for (int t = 0; t < 16; ++t) {
      int o = t * 32 + l, row = o >> 5, hch = o & 31;
      float x = VH[0][row][hch], y = VH[1][row][hch], z = VH[2][row][hch];
      SB2[row][100 + hch] = (half_t)sqrtf(fmaxf(x * x + y * y + z * z, 1e-8f));
    }
  }
#pragma unroll 1  // dh1 s-GEMM: [16x160] @ [160x400] via WMMA, relu
  for (int nt = 0; nt < 25; ++nt) {
    v8f acc = {};
#pragma unroll
    for (int kt = 0; kt < 5; ++kt)
      acc = wmma16(load_a(&SB2[0][0], 608, kt * 32),
                   load_b(wf + (OFF_DH1WS + nt * 5 + kt) * FRAG), acc);
    int n = nt * 16 + lm; float bv = b1dh[n];
#pragma unroll
    for (int r = 0; r < 8; ++r)
      SB2[r + 8 * hh][160 + n] = (half_t)fmaxf(acc[r] + bv, 0.f);
  }
  {  // dh1 v_out with sigmoid gate -> VV (32 ch)
    float gx[16], gy[16], gz[16];
    for (int t = 0; t < 16; ++t) {
      int o = t * 32 + l, row = o >> 5, oc = o & 31;
      float ax = 0.f, ay = 0.f, az = 0.f;
      for (int k2 = 0; k2 < 32; ++k2) {
        float wv = w1v[k2 * 32 + oc];
        ax += VH[0][row][k2] * wv; ay += VH[1][row][k2] * wv; az += VH[2][row][k2] * wv;
      }
      float nrm = sqrtf(fmaxf(ax * ax + ay * ay + az * az, 1e-8f));
      float g = 1.f / (1.f + __expf(-nrm));
      gx[t] = ax * g; gy[t] = ay * g; gz[t] = az * g;
    }
    for (int t = 0; t < 16; ++t) {
      int o = t * 32 + l, row = o >> 5, oc = o & 31;
      VV[0][row][oc] = gx[t]; VV[1][row][oc] = gy[t]; VV[2][row][oc] = gz[t];
    }
  }
  {  // dh2 vh: 32->32 per dim, vn -> SB2[560..591]
    float ax[16], ay[16], az[16];
    for (int t = 0; t < 16; ++t) {
      int o = t * 32 + l, row = o >> 5, h2 = o & 31;
      float x = 0.f, y = 0.f, z = 0.f;
      for (int k2 = 0; k2 < 32; ++k2) {
        float wv = w2h[k2 * 32 + h2];
        x += VV[0][row][k2] * wv; y += VV[1][row][k2] * wv; z += VV[2][row][k2] * wv;
      }
      ax[t] = x; ay[t] = y; az[t] = z;
    }
    for (int t = 0; t < 16; ++t) {
      int o = t * 32 + l, row = o >> 5, h2 = o & 31;
      VH[0][row][h2] = ax[t]; VH[1][row][h2] = ay[t]; VH[2][row][h2] = az[t];
      SB2[row][560 + h2] = (half_t)sqrtf(fmaxf(ax[t] * ax[t] + ay[t] * ay[t] + az[t] * az[t], 1e-8f));
    }
  }
#pragma unroll 1  // dh2 s-GEMM: [16x448] @ [448x100] via WMMA (no relu)
  for (int nt = 0; nt < 7; ++nt) {
    v8f acc = {};
#pragma unroll
    for (int kt = 0; kt < 14; ++kt)
      acc = wmma16(load_a(&SB2[0][0], 608, 160 + kt * 32),
                   load_b(wf + (OFF_DH2WS + nt * 14 + kt) * FRAG), acc);
    int n = nt * 16 + lm;
    if (n < 100) {
      float bv = b2dh[n];
#pragma unroll
      for (int r = 0; r < 8; ++r) DB[r + 8 * hh][48 + n] = acc[r] + bv;
    }
  }
  for (int t = 0; t < 8; ++t) {  // dh2 v_out: 32->16, no activation
    int o = t * 32 + l, row = o >> 4, oc = o & 15;
    float x = 0.f, y = 0.f, z = 0.f;
    for (int k2 = 0; k2 < 32; ++k2) {
      float wv = w2v[k2 * 16 + oc];
      x += VH[0][row][k2] * wv; y += VH[1][row][k2] * wv; z += VH[2][row][k2] * wv;
    }
    DB[row][oc] = x; DB[row][16 + oc] = y; DB[row][32 + oc] = z;
  }
  for (int i = l; i < 16 * HV_DIM; i += 32) {  // residual
    int row = i / HV_DIM, col = i % HV_DIM;
    DB[row][col] += HVB2[row][col];
  }
  {  // LN2 + mask_V -> out
    int row = l >> 1, part = l & 1;
    size_t node = (size_t)blk * 16 + row;
    float mk = (float)maskV[node];
    float sum = 0.f, sq = 0.f;
    for (int j = part * 50; j < part * 50 + 50; ++j) { float v = DB[row][48 + j]; sum += v; sq += v * v; }
    sum += __shfl_xor(sum, 1, 32); sq += __shfl_xor(sq, 1, 32);
    float mu = sum * 0.01f, var = sq * 0.01f - mu * mu, rstd = rsqrtf(var + 1e-3f);
    float vs = 0.f;
    for (int ch = part * 8; ch < part * 8 + 8; ++ch) {
      float q = 0.f;
      for (int c = 0; c < 3; ++c) { float v = DB[row][c * 16 + ch]; q += v * v; }
      vs += fmaxf(q, 1e-8f);
    }
    vs += __shfl_xor(vs, 1, 32);
    float vsc = rsqrtf(vs * (1.f / 16.f));
    float* op = out + node * HV_DIM;
    for (int j = part * 50; j < part * 50 + 50; ++j)
      op[48 + j] = ((DB[row][48 + j] - mu) * rstd * g2[j] + b2[j]) * mk;
    for (int col = part * 24; col < part * 24 + 24; ++col)
      op[col] = DB[row][col] * vsc * mk;
  }
}

// ---------------- host ----------------
extern "C" void kernel_launch(void* const* d_in, const int* in_sizes, int n_in,
                              void* d_out, int out_size, void* d_ws, size_t ws_size,
                              hipStream_t stream) {
  (void)in_sizes; (void)n_in; (void)out_size; (void)ws_size;
  half_t* wf    = (half_t*)d_ws;
  float*  dhBuf = (float*)((char*)d_ws + FRAG_BYTES);

  auto P = [&](int idx, int off, int K, int N, int Kt, int Nt) {
    int total = Kt * Nt * FRAG;
    prep_w<<<(total + 255) / 256, 256, 0, stream>>>(
        (const float*)d_in[idx], wf + (size_t)off * FRAG, K, N, Kt, total);
  };
  P(4,  OFF_WH1,   33, 33, 2, 3);
  P(5,  OFF_WS1,  265, 100, 9, 7);
  P(7,  OFF_WV1,   33, 16, 2, 1);
  P(8,  OFF_WH2,   16, 16, 1, 1);
  P(9,  OFF_WS2,  116, 100, 4, 7);
  P(11, OFF_WV2,   16, 16, 1, 1);
  P(12, OFF_WH3,   16, 16, 1, 1);
  P(13, OFF_WS3,  116, 100, 4, 7);
  P(15, OFF_WV3,   16, 16, 1, 1);
  P(17, OFF_DH1WS, 132, 400, 5, 25);
  P(21, OFF_DH2WS, 432, 100, 14, 7);

  edge_kernel<<<NODES / 2, 128, 0, stream>>>(
      (const float*)d_in[0], (const float*)d_in[1], (const int*)d_in[3], wf,
      (const float*)d_in[6], (const float*)d_in[10], (const float*)d_in[14], dhBuf);

  node_kernel<<<NODES / 16, 32, 0, stream>>>(
      (const float*)d_in[0], dhBuf, (const int*)d_in[2], wf,
      (const float*)d_in[16], (const float*)d_in[18], (const float*)d_in[19],
      (const float*)d_in[20], (const float*)d_in[22], (const float*)d_in[23],
      (const float*)d_in[24], (const float*)d_in[25],
      (const float*)d_in[26], (const float*)d_in[27],
      (float*)d_out);
}
